// SepFlexConv_31387620999360
// MI455X (gfx1250) — matrix operand order repaired and checked by reading, not compiled, not run on hardware
//
#include <hip/hip_runtime.h>

// ---------------- problem constants ----------------
#define BB   16
#define CC   384
#define LL   4096
#define KK   4096
#define KH   32
#define NLAY 3
#define HID  384
#define OMEGA0 976.0f

#define LP   4128                 // padded x row (L + 32 zeros), 16B-aligned rows
#define GP   4160                 // reversed-kernel row length (4096 + zero pad)

typedef __attribute__((ext_vector_type(16))) __bf16        v16bf;
typedef __attribute__((ext_vector_type(8)))  float         v8f;
typedef __attribute__((ext_vector_type(4)))  unsigned int  u32x4;
typedef __attribute__((ext_vector_type(4)))  int           i32x4;
typedef __attribute__((ext_vector_type(8)))  int           i32x8;

#if defined(__has_builtin)
#if __has_builtin(__builtin_amdgcn_tensor_load_to_lds) && __has_builtin(__builtin_amdgcn_s_wait_tensorcnt)
#define HAVE_TDM 1
#endif
#endif
#ifndef HAVE_TDM
#define HAVE_TDM 0
#endif

// ---------------- workspace layout (bytes) ----------------
#define X_BYTES   ((size_t)BB * CC * LP * 2)            // 50,724,864
#define G_BYTES   ((size_t)CC * GP * 2)                 //  3,194,880 (each copy)
#define PW_BYTES  ((size_t)HID * CC * 2)                //    294,912
#define OFF_G0    (X_BYTES)
#define OFF_G1    (OFF_G0 + G_BYTES)
#define OFF_PW    (OFF_G1 + G_BYTES)
#define OFF_Y     (OFF_PW + PW_BYTES)

#define CONV_SMEM ((16 * LP + 2 * GP) * 2)              // 148,736 B < 320 KB/WGP

#if HAVE_TDM
// ---------------- Tensor Data Mover: 2D tile global -> LDS ----------------
__device__ __forceinline__ void tdm_load_2d(unsigned lds_off, const void* gptr,
                                            unsigned tile_d0, unsigned tile_d1,
                                            unsigned long long stride_d0,
                                            bool pad /* +4 dwords per 64 dwords */) {
  unsigned long long ga = (unsigned long long)(uintptr_t)gptr;
  u32x4 g0;
  g0[0] = 1u;                                            // count=1, user D#
  g0[1] = lds_off;                                       // lds_addr (bytes)
  g0[2] = (unsigned)(ga & 0xffffffffull);                // global_addr[31:0]
  g0[3] = (unsigned)((ga >> 32) & 0x01ffffffull) | (2u << 30);   // [56:32] | type=2
  unsigned w0 = 0x00010000u;                             // data_size=1 -> 2 bytes
  if (pad) w0 |= (1u << 20) | (5u << 22) | (3u << 25);   // pad: every 64 dw, +4 dw
  i32x8 g1;
  g1[0] = (int)w0;
  g1[1] = (int)((tile_d0 & 0xffffu) << 16);              // tensor_dim0 lo16
  g1[2] = (int)((tile_d0 >> 16) | ((tile_d1 & 0xffffu) << 16)); // dim0 hi | dim1 lo
  g1[3] = (int)((tile_d1 >> 16) | (tile_d0 << 16));      // dim1 hi | tile_dim0
  g1[4] = (int)(tile_d1 & 0xffffu);                      // tile_dim1 (tile_dim2=0)
  g1[5] = (int)(unsigned)(stride_d0 & 0xffffffffull);    // dim0_stride lo32
  g1[6] = (int)(unsigned)((stride_d0 >> 32) & 0xffffull);// dim0_stride hi16
  g1[7] = 0;
  i32x4 z4 = {0, 0, 0, 0};
#if __clang_major__ >= 23
  i32x8 z8 = {0, 0, 0, 0, 0, 0, 0, 0};
  __builtin_amdgcn_tensor_load_to_lds(g0, g1, z4, z4, z8, 0);
#else
  __builtin_amdgcn_tensor_load_to_lds(g0, g1, z4, z4, 0);
#endif
}
#endif // HAVE_TDM

// ============================================================
// Stage 0a: zero the reversed-kernel buffers (pads must be 0)
// ============================================================
__global__ void k_zero(uint4* p, size_t n4) {
  size_t i = (size_t)blockIdx.x * blockDim.x + threadIdx.x;
  if (i < n4) p[i] = make_uint4(0u, 0u, 0u, 0u);
}

// ============================================================
// Stage 0b: x fp32 -> bf16 with zero-padded rows [B*C][LP]
// ============================================================
__global__ void k_cvt_x(const float* __restrict__ x, __bf16* __restrict__ xb) {
  size_t i = (size_t)blockIdx.x * blockDim.x + threadIdx.x;
  size_t total = (size_t)BB * CC * LP;
  if (i >= total) return;
  size_t row = i / LP;
  int    l   = (int)(i - row * LP);
  xb[i] = (l < LL) ? (__bf16)x[row * LL + l] : (__bf16)0.0f;
}

// ============================================================
// Stage 0c: pw_w fp32 -> bf16 [HID][C]
// ============================================================
__global__ void k_cvt_pw(const float* __restrict__ pw, __bf16* __restrict__ pwb) {
  int i = blockIdx.x * blockDim.x + threadIdx.x;
  if (i < HID * CC) pwb[i] = (__bf16)pw[i];
}

// ============================================================
// Stage 1: Gabor-MFN kernel generation -> reversed, masked,
// scaled, bf16 kernel g0[c][j]=kern[c][4095-j], g1[j]=g0[j+1]
// ============================================================
__global__ __launch_bounds__(256) void k_gen(
    const float* __restrict__ mask_mean, const float* __restrict__ mask_sigma,
    const float* __restrict__ gfw, const float* __restrict__ gfb,
    const float* __restrict__ gga, const float* __restrict__ gmu,
    const float* __restrict__ lw,  const float* __restrict__ lb,
    const float* __restrict__ ow,  const float* __restrict__ ob,
    __bf16* __restrict__ g0, __bf16* __restrict__ g1) {
  int p = blockIdx.x * 256 + threadIdx.x;
  if (p >= KK) return;
  float pos = -1.0f + 2.0f * (float)p / (float)(KK - 1);

  float h[KH], nh[KH];
  #pragma unroll
  for (int j = 0; j < KH; ++j) {                 // gabor filter 0
    float d = pos - gmu[j];
    float e = __expf(-0.5f * gga[j] * gga[j] * d * d);
    h[j] = e * __sinf(OMEGA0 * (pos * gfw[j]) + gfb[j]);
  }
  for (int i = 0; i < NLAY; ++i) {               // MFN hidden layers
    const float* W  = lw  + i * KH * KH;
    const float* Bv = lb  + i * KH;
    const float* mu = gmu + (i + 1) * KH;
    const float* ga = gga + (i + 1) * KH;
    const float* fw = gfw + (i + 1) * KH;
    const float* fb = gfb + (i + 1) * KH;
    for (int j = 0; j < KH; ++j) {
      float s = Bv[j];
      #pragma unroll
      for (int k = 0; k < KH; ++k) s += W[j * KH + k] * h[k];
      float d = pos - mu[j];
      float e = __expf(-0.5f * ga[j] * ga[j] * d * d);
      nh[j] = s * e * __sinf(OMEGA0 * (pos * fw[j]) + fb[j]);
    }
    #pragma unroll
    for (int j = 0; j < KH; ++j) h[j] = nh[j];
  }
  float z = (pos - mask_mean[0]) / mask_sigma[0];
  float scale = __expf(-0.5f * z * z) * rsqrtf((float)CC * (float)KK);

  int j0 = (KK - 1) - p;                         // reversed index
  for (int c = 0; c < CC; ++c) {
    float s = ob[c];
    #pragma unroll
    for (int j = 0; j < KH; ++j) s += ow[c * KH + j] * h[j];
    __bf16 v = (__bf16)(s * scale);
    g0[(size_t)c * GP + j0] = v;
    if (j0 >= 1) g1[(size_t)c * GP + j0 - 1] = v;   // g1[j] = g0[j+1]
  }
}

// ============================================================
// Stage 2: causal depthwise conv as Toeplitz WMMA (bf16).
// One wave owns 4 l-tiles spaced 1024 apart; the Toeplitz B
// fragment depends only on d = l0 - m0, so one B feeds 4 WMMAs
// (DS/WMMA: 10 -> 4). Tiles retire from the d-loop one by one
// via uniform SALU bounds -- EXEC stays all-ones throughout.
// ============================================================
union AU { uint4 q[2]; v16bf v; };
union BU { unsigned int u[8]; v16bf v; };

__device__ __forceinline__ void loadA(const __bf16* xrow, int m0, int hi, AU& a) {
  const int ab = m0 + (hi << 4);
  a.q[0] = *(const uint4*)(xrow + ab);
  a.q[1] = *(const uint4*)(xrow + ab + 8);
}
__device__ __forceinline__ void loadB(const __bf16* gl0, const __bf16* gl1,
                                      int d, int n, int hi, BU& b) {
  const int j0 = (KK - 1) - d - n + (hi << 4);
  const unsigned int* gu = (j0 & 1) ? (const unsigned int*)(gl1 + (j0 - 1))
                                    : (const unsigned int*)(gl0 + j0);
  #pragma unroll
  for (int q = 0; q < 8; ++q) b.u[q] = gu[q];
}

__global__ __launch_bounds__(256) void k_conv(
    const __bf16* __restrict__ xb, const __bf16* __restrict__ g0g,
    const __bf16* __restrict__ g1g, __bf16* __restrict__ yb) {
  extern __shared__ char smem_raw[];
  __bf16* xl  = (__bf16*)smem_raw;       // [16][LP]
  __bf16* gl0 = xl + 16 * LP;            // [GP]
  __bf16* gl1 = gl0 + GP;                // [GP]

  const int c   = blockIdx.x;
  const int tid = threadIdx.x;
  const int wU  = __builtin_amdgcn_readfirstlane(tid >> 5);   // uniform wave id

#if HAVE_TDM
  if (wU == 0) {   // one wave drives the Tensor Data Mover
    tdm_load_2d((unsigned)(uintptr_t)xl,  xb  + (size_t)c * LP, LP, 16,
                (unsigned long long)CC * LP, false);
    tdm_load_2d((unsigned)(uintptr_t)gl0, g0g + (size_t)c * GP, GP, 1, GP, false);
    tdm_load_2d((unsigned)(uintptr_t)gl1, g1g + (size_t)c * GP, GP, 1, GP, false);
    __builtin_amdgcn_s_wait_tensorcnt(0);
  }
#else
  {
    const int r4 = LP / 8;                              // uint4 per row
    for (int i = tid; i < 16 * r4; i += 256) {
      int row = i / r4, col = i - row * r4;
      ((uint4*)(xl + row * LP))[col] =
          ((const uint4*)(xb + ((size_t)(row * CC + c)) * LP))[col];
    }
    const int g4 = GP / 8;
    for (int i = tid; i < g4; i += 256) {
      ((uint4*)gl0)[i] = ((const uint4*)(g0g + (size_t)c * GP))[i];
      ((uint4*)gl1)[i] = ((const uint4*)(g1g + (size_t)c * GP))[i];
    }
  }
#endif
  __syncthreads();

  const int lane = tid & 31;
  const int n    = lane & 15;      // B col (output position) / A row (batch)
  const int hi   = lane >> 4;
  const __bf16* xrow = xl + n * LP;

  for (int j = 0; j < 8; ++j) {            // 8 quads of 4 tiles per wave
    const int q    = wU + (j << 3);        // quad id 0..63 (uniform)
    const int l0   = q << 4;
    const int l1   = l0 + 1024;
    const int l2   = l0 + 2048;
    const int l3   = l0 + 3072;
    const int dmin = l0 & 16;              // all 4 tiles share l0 mod 32
    v8f acc0 = {0.f,0.f,0.f,0.f,0.f,0.f,0.f,0.f};
    v8f acc1 = {0.f,0.f,0.f,0.f,0.f,0.f,0.f,0.f};
    v8f acc2 = {0.f,0.f,0.f,0.f,0.f,0.f,0.f,0.f};
    v8f acc3 = {0.f,0.f,0.f,0.f,0.f,0.f,0.f,0.f};

    int d = dmin;
    for (; d <= l0; d += 32) {             // all 4 tiles active: 1 B : 4 WMMA
      BU bfr; loadB(gl0, gl1, d, n, hi, bfr);
      AU a0, a1, a2, a3;
      loadA(xrow, l0 - d, hi, a0);
      loadA(xrow, l1 - d, hi, a1);
      loadA(xrow, l2 - d, hi, a2);
      loadA(xrow, l3 - d, hi, a3);
      acc0 = __builtin_amdgcn_wmma_f32_16x16x32_bf16(false, a0.v, false, bfr.v, (short)0, acc0, false, false);
      acc1 = __builtin_amdgcn_wmma_f32_16x16x32_bf16(false, a1.v, false, bfr.v, (short)0, acc1, false, false);
      acc2 = __builtin_amdgcn_wmma_f32_16x16x32_bf16(false, a2.v, false, bfr.v, (short)0, acc2, false, false);
      acc3 = __builtin_amdgcn_wmma_f32_16x16x32_bf16(false, a3.v, false, bfr.v, (short)0, acc3, false, false);
    }
    for (; d <= l1; d += 32) {             // tiles 1..3 active
      BU bfr; loadB(gl0, gl1, d, n, hi, bfr);
      AU a1, a2, a3;
      loadA(xrow, l1 - d, hi, a1);
      loadA(xrow, l2 - d, hi, a2);
      loadA(xrow, l3 - d, hi, a3);
      acc1 = __builtin_amdgcn_wmma_f32_16x16x32_bf16(false, a1.v, false, bfr.v, (short)0, acc1, false, false);
      acc2 = __builtin_amdgcn_wmma_f32_16x16x32_bf16(false, a2.v, false, bfr.v, (short)0, acc2, false, false);
      acc3 = __builtin_amdgcn_wmma_f32_16x16x32_bf16(false, a3.v, false, bfr.v, (short)0, acc3, false, false);
    }
    for (; d <= l2; d += 32) {             // tiles 2..3 active
      BU bfr; loadB(gl0, gl1, d, n, hi, bfr);
      AU a2, a3;
      loadA(xrow, l2 - d, hi, a2);
      loadA(xrow, l3 - d, hi, a3);
      acc2 = __builtin_amdgcn_wmma_f32_16x16x32_bf16(false, a2.v, false, bfr.v, (short)0, acc2, false, false);
      acc3 = __builtin_amdgcn_wmma_f32_16x16x32_bf16(false, a3.v, false, bfr.v, (short)0, acc3, false, false);
    }
    for (; d <= l3; d += 32) {             // tile 3 only
      BU bfr; loadB(gl0, gl1, d, n, hi, bfr);
      AU a3;
      loadA(xrow, l3 - d, hi, a3);
      acc3 = __builtin_amdgcn_wmma_f32_16x16x32_bf16(false, a3.v, false, bfr.v, (short)0, acc3, false, false);
    }
    // D: VGPR r -> batch row r (+8 for lanes 16-31), col n
    #pragma unroll
    for (int r = 0; r < 8; ++r) {
      int bo = r + (hi << 3);
      size_t base = ((size_t)(bo * CC + c)) << 12;
      yb[base + l0 + n] = (__bf16)acc0[r];
      yb[base + l1 + n] = (__bf16)acc1[r];
      yb[base + l2 + n] = (__bf16)acc2[r];
      yb[base + l3 + n] = (__bf16)acc3[r];
    }
  }
}

// ============================================================
// Stage 3: pointwise conv = GEMM out[o, bl] = pw[o,c] y[c, bl]
// WMMA bf16, K = 384 in 12 chunks. Each wave computes TWO
// o-tiles sharing the expensive strided B fragment (DS/WMMA
// halves to 8). TDM-staged (HW-padded rows) or manual fallback.
// ============================================================
__global__ __launch_bounds__(256) void k_pw(
    const __bf16* __restrict__ pwb, const __bf16* __restrict__ yb,
    float* __restrict__ out) {
  __shared__ __align__(16) __bf16 yl[32 * 136];     // stride 136: 16B-aligned rows

  const int lsup = blockIdx.x;           // 32 l-superblocks of 128
  const int ot   = blockIdx.y;           // 12 o-pairs (2x16 rows each)
  const int b    = blockIdx.z;           // 16 batches
  const int tid  = threadIdx.x;
  const int lane = tid & 31;
  const int wU   = __builtin_amdgcn_readfirstlane(tid >> 5);  // 8 l-subtiles
  const int lblk = lsup * 128;
  const int l0   = lblk + (wU << 4);
  const int o0   = ot << 5;
  const int n    = lane & 15;
  const int hi   = lane >> 4;

  v8f accA = {0.f,0.f,0.f,0.f,0.f,0.f,0.f,0.f};
  v8f accB = {0.f,0.f,0.f,0.f,0.f,0.f,0.f,0.f};

  for (int c0 = 0; c0 < CC; c0 += 32) {
    __syncthreads();
#if HAVE_TDM
    if (wU == 0) {   // 32x128 bf16 tile; HW pad -> 136-half LDS rows
      tdm_load_2d((unsigned)(uintptr_t)yl,
                  yb + (((size_t)(b * CC + c0)) << 12) + lblk,
                  128, 32, (unsigned long long)LL, true);
      __builtin_amdgcn_s_wait_tensorcnt(0);
    }
#else
    for (int i = tid; i < 512; i += 256) {
      int row = i >> 4, col = i & 15;
      ((uint4*)(yl + row * 136))[col] =
          ((const uint4*)(yb + (((size_t)(b * CC + c0 + row)) << 12) + lblk))[col];
      if (c0 + 32 < CC)   // prefetch next K-chunk tile (global_prefetch_b8)
        __builtin_prefetch(yb + (((size_t)(b * CC + c0 + 32 + row)) << 12) + lblk, 0, 0);
    }
#endif
    __syncthreads();

    union AUL { uint4 q[2]; v16bf v; } a0, a1;
    const int ab = c0 + (hi << 3);
    a0.q[0] = *(const uint4*)(pwb + (o0 + n) * CC + ab);
    a0.q[1] = *(const uint4*)(pwb + (o0 + n) * CC + ab + 16);
    a1.q[0] = *(const uint4*)(pwb + (o0 + 16 + n) * CC + ab);
    a1.q[1] = *(const uint4*)(pwb + (o0 + 16 + n) * CC + ab + 16);
    if (c0 + 32 < CC)
      __builtin_prefetch(pwb + (o0 + n) * CC + c0 + 32, 0, 3);

    union BUS { unsigned short s[16]; v16bf v; } bf;
    #pragma unroll
    for (int qq = 0; qq < 16; ++qq) {
      int kk = (hi << 3) + (qq & 7) + ((qq >> 3) << 4);
      bf.s[qq] = *(const unsigned short*)(yl + kk * 136 + (wU << 4) + n);
    }
    accA = __builtin_amdgcn_wmma_f32_16x16x32_bf16(false, a0.v, false, bf.v, (short)0, accA, false, false);
    accB = __builtin_amdgcn_wmma_f32_16x16x32_bf16(false, a1.v, false, bf.v, (short)0, accB, false, false);
  }

  #pragma unroll
  for (int r = 0; r < 8; ++r) {
    int oo = o0 + r + (hi << 3);
    out[(((size_t)(b * HID + oo)) << 12) + l0 + n] = accA[r];
    out[(((size_t)(b * HID + oo + 16)) << 12) + l0 + n] = accB[r];
  }
}

// ============================================================
extern "C" void kernel_launch(void* const* d_in, const int* in_sizes, int n_in,
                              void* d_out, int out_size, void* d_ws, size_t ws_size,
                              hipStream_t stream) {
  const float* x         = (const float*)d_in[0];
  const float* mask_mean = (const float*)d_in[1];
  const float* mask_sig  = (const float*)d_in[2];
  const float* gfw       = (const float*)d_in[3];
  const float* gfb       = (const float*)d_in[4];
  const float* gga       = (const float*)d_in[5];
  const float* gmu       = (const float*)d_in[6];
  const float* lw        = (const float*)d_in[7];
  const float* lb        = (const float*)d_in[8];
  const float* ow        = (const float*)d_in[9];
  const float* ob        = (const float*)d_in[10];
  const float* pw        = (const float*)d_in[11];
  (void)in_sizes; (void)n_in; (void)out_size; (void)ws_size;

  char*   ws  = (char*)d_ws;
  __bf16* xb  = (__bf16*)(ws);
  __bf16* g0  = (__bf16*)(ws + OFF_G0);
  __bf16* g1  = (__bf16*)(ws + OFF_G1);
  __bf16* pwb = (__bf16*)(ws + OFF_PW);
  __bf16* yb  = (__bf16*)(ws + OFF_Y);

  // Stage 0: zero kernel buffers (pads encode causal zero taps)
  {
    size_t n4 = (2 * G_BYTES) / 16;
    k_zero<<<(unsigned)((n4 + 255) / 256), 256, 0, stream>>>((uint4*)g0, n4);
  }
  // Stage 0: precision conversion
  {
    size_t total = (size_t)BB * CC * LP;
    k_cvt_x<<<(unsigned)((total + 255) / 256), 256, 0, stream>>>(x, xb);
    k_cvt_pw<<<(HID * CC + 255) / 256, 256, 0, stream>>>(pw, pwb);
  }
  // Stage 1: MFN kernel generation (reversed + masked + scaled)
  k_gen<<<KK / 256, 256, 0, stream>>>(mask_mean, mask_sig, gfw, gfb, gga, gmu,
                                      lw, lb, ow, ob, g0, g1);
  // Stage 2: depthwise causal conv via Toeplitz WMMA (149 KB LDS / WGP)
  (void)hipFuncSetAttribute((const void*)k_conv,
                            hipFuncAttributeMaxDynamicSharedMemorySize,
                            (int)CONV_SMEM);
  k_conv<<<CC, 256, CONV_SMEM, stream>>>(xb, g0, g1, yb);
  // Stage 3: pointwise GEMM via WMMA
  k_pw<<<dim3(32, HID / 32, BB), 256, 0, stream>>>(pwb, yb, (float*)d_out);
}
